// GATAgent_23733989277999
// MI455X (gfx1250) — matrix operand order repaired and checked by reading, not compiled, not run on hardware
//
#include <hip/hip_runtime.h>
#include <hip/hip_bf16.h>

#define BATCH 8192
#define NNODE 32
#define DIN   128
#define HDIM  64
#define ADIM  16
#define WAVES 4

typedef _Float16 half_t;
typedef __attribute__((ext_vector_type(16))) _Float16 v16h;
typedef __attribute__((ext_vector_type(8)))  _Float16 v8h;
typedef __attribute__((ext_vector_type(8)))  float    v8f;
typedef unsigned int u32x4 __attribute__((ext_vector_type(4)));
typedef int          i32x4 __attribute__((ext_vector_type(4)));
typedef int          i32x8 __attribute__((ext_vector_type(8)));

static __device__ inline v16h pack16(v8h lo, v8h hi) {
    v16h r;
#pragma unroll
    for (int j = 0; j < 8; ++j) { r[j] = lo[j]; r[8 + j] = hi[j]; }
    return r;
}

static __device__ inline v8f wmma_f16(v16h a, v16h b, v8f c) {
    return __builtin_amdgcn_wmma_f32_16x16x32_f16(
        /*neg_a=*/false, a, /*neg_b=*/false, b,
        /*c_mod=*/(short)0, c, /*reuse_a=*/false, /*reuse_b=*/false);
}

// A operand (16x32 f16) from row-major f16 LDS buffer, leading dim ld (halves).
static __device__ inline v16h ldsA(const half_t* buf, int ld, int m0, int k0, int lane) {
    int r  = m0 + (lane & 15);
    int kb = k0 + ((lane & 16) ? 16 : 0);
    const half_t* p = buf + r * ld + kb;
    return pack16(*(const v8h*)p, *(const v8h*)(p + 8));
}

// A operand directly from row-major f32 global buffer (converted to f16).
static __device__ inline v16h gA_f32(const float* base, int ld, int m0, int k0, int lane) {
    int r  = m0 + (lane & 15);
    int kb = k0 + ((lane & 16) ? 16 : 0);
    const float* p = base + (size_t)r * ld + kb;
    v16h a;
#pragma unroll
    for (int j = 0; j < 8; ++j) { a[j] = (half_t)p[j]; a[8 + j] = (half_t)p[8 + j]; }
    return a;
}

// B operand (32x16) where the source is stored [n][k] row-major f32 (W matrices: B = W^T).
static __device__ inline v16h gB_W(const float* W, int ldk, int n0, int k0, int lane) {
    int n  = n0 + (lane & 15);
    int kb = k0 + ((lane & 16) ? 16 : 0);
    const float* p = W + (size_t)n * ldk + kb;
    v16h b;
#pragma unroll
    for (int j = 0; j < 16; ++j) b[j] = (half_t)p[j];
    return b;
}

// B operand from f16 LDS stored [n][k] (k contiguous per lane).
static __device__ inline v16h ldsB_nk(const half_t* buf, int ldk, int n0, int k0, int lane) {
    int n  = n0 + (lane & 15);
    int kb = k0 + ((lane & 16) ? 16 : 0);
    const half_t* p = buf + n * ldk + kb;
    return pack16(*(const v8h*)p, *(const v8h*)(p + 8));
}

// B operand from f16 LDS stored [k][n] (strided gather along k).
static __device__ inline v16h ldsB_kn(const half_t* buf, int ldn, int n0, int k0, int lane) {
    int n  = n0 + (lane & 15);
    int kb = k0 + ((lane & 16) ? 16 : 0);
    v16h b;
#pragma unroll
    for (int j = 0; j < 16; ++j) b[j] = buf[(kb + j) * ldn + n];
    return b;
}

__global__ __launch_bounds__(128, 1)
void gat_gru_fused(const float* __restrict__ inp,  const float* __restrict__ hidden,
                   const float* __restrict__ adj,
                   const float* __restrict__ w1,   const float* __restrict__ b1,
                   const float* __restrict__ w2,   const float* __restrict__ b2,
                   const float* __restrict__ w3,   const float* __restrict__ b3,
                   const float* __restrict__ Wih,  const float* __restrict__ Whh,
                   const float* __restrict__ bih,  const float* __restrict__ bhh,
                   const float* __restrict__ Wq,   const float* __restrict__ bq,
                   float* __restrict__ outq,       float* __restrict__ outh)
{
    __shared__ __align__(16) half_t sW1T[HDIM * DIN];            // 16 KB, w1^T as [n][k] f16
    __shared__ __align__(16) half_t sXA[WAVES][NNODE * HDIM];    // 16 KB, x then a (aliased)
    __shared__ __align__(16) half_t sH [WAVES][NNODE * HDIM];    // 16 KB, h_in then h
    __shared__ __align__(16) half_t sAw[WAVES][NNODE * NNODE];   //  8 KB, softmax weights
    __shared__ float sF2[WAVES][NNODE];
    __shared__ float sF3[WAVES][NNODE];
    __shared__ __align__(16) float sAdj[NNODE * NNODE];          //  4 KB, filled by TDM

    const int tid  = threadIdx.x;
    const int lane = tid & 31;
    const int wv   = tid >> 5;
    const long long b = (long long)blockIdx.x * WAVES + wv;

    // ---- Tensor Data Mover: adj (32x32 f32) -> LDS, issued by wave 0 ----
    if (wv == 0) {
        unsigned long long ga = (unsigned long long)(uintptr_t)adj;
        unsigned int la = (unsigned int)(uintptr_t)(&sAdj[0]);   // low 32 bits of flat == LDS byte addr
        u32x4 g0;
        g0.x = 1u;                                                // count=1 (valid descriptor)
        g0.y = la;                                                // lds_addr
        g0.z = (unsigned int)(ga & 0xffffffffull);                // global_addr[31:0]
        g0.w = (unsigned int)((ga >> 32) & 0x1ffffffull) | (2u << 30); // addr[56:32] | type=2
        i32x8 g1;
        g1[0] = 0x00020000;            // data_size = 4B
        g1[1] = (int)(32u << 16);      // tensor_dim0 = 32
        g1[2] = (int)(32u << 16);      // tensor_dim1 = 32
        g1[3] = (int)(32u << 16);      // tile_dim0 = 32
        g1[4] = 32;                    // tile_dim1 = 32
        g1[5] = 32;                    // tensor_dim0_stride = 32
        g1[6] = (int)(1024u << 16);    // tensor_dim1_stride = 1024
        g1[7] = 0;
        i32x4 gz = {0, 0, 0, 0};
#if defined(__has_include) && __has_include(<hip/amd_detail/amd_gfx1250_TDM.h>)
        i32x8 gz8 = {0, 0, 0, 0, 0, 0, 0, 0};
        __builtin_amdgcn_tensor_load_to_lds(g0, g1, gz, gz, gz8, 0);
#else
        __builtin_amdgcn_tensor_load_to_lds(g0, g1, gz, gz, 0);
#endif
    }

    // ---- Stage w1^T into LDS as f16 (shared by all 4 waves) ----
#pragma unroll 4
    for (int t = tid; t < DIN * HDIM; t += 128) {
        int k = t >> 6, n = t & 63;
        sW1T[n * DIN + k] = (half_t)w1[t];
    }

    // ---- Stage h_in (one row per lane, contiguous 256B reads) ----
    {
        const float* hr = hidden + (b * NNODE + lane) * HDIM;
        half_t* dst = &sH[wv][lane * HDIM];
#pragma unroll
        for (int k = 0; k < HDIM; ++k) dst[k] = (half_t)hr[k];
    }

    if (b + WAVES < BATCH)
        __builtin_prefetch(inp + (size_t)(b + WAVES) * (NNODE * DIN), 0, 1);

    if (wv == 0) __builtin_amdgcn_s_wait_tensorcnt(0);
    __syncthreads();

    // ================= GEMM1: x = relu(In[32x128] @ w1[128x64] + b1) =================
    const float* inB = inp + b * (long long)(NNODE * DIN);
    half_t* xbuf = &sXA[wv][0];
    const v8f z8 = {0.f, 0.f, 0.f, 0.f, 0.f, 0.f, 0.f, 0.f};
#pragma unroll
    for (int mi = 0; mi < 2; ++mi) {
#pragma unroll
        for (int ni = 0; ni < 4; ++ni) {
            v8f acc = z8;
#pragma unroll
            for (int kk = 0; kk < 4; ++kk) {
                v16h a  = gA_f32(inB, DIN, mi * 16, kk * 32, lane);
                v16h bb = ldsB_nk(sW1T, DIN, ni * 16, kk * 32, lane);
                acc = wmma_f16(a, bb, acc);
            }
            int col = ni * 16 + (lane & 15);
            float bias = b1[col];
            int r0 = mi * 16 + ((lane & 16) ? 8 : 0);
#pragma unroll
            for (int r = 0; r < 8; ++r) {
                float v = acc[r] + bias;
                v = v > 0.f ? v : 0.f;
                xbuf[(r0 + r) * HDIM + col] = (half_t)v;
            }
        }
    }
    asm volatile("s_wait_dscnt 0x0" ::: "memory");

    // ================= f2/f3: per-row dots (lane == row) =================
    {
        const half_t* xr = xbuf + lane * HDIM;
        float f2 = b2[0], f3 = b3[0];
#pragma unroll
        for (int k = 0; k < HDIM; ++k) {
            float xv = (float)xr[k];
            f2 += xv * w2[k];
            f3 += xv * w3[k];
        }
        sF2[wv][lane] = f2;
        sF3[wv][lane] = f3;
    }
    asm volatile("s_wait_dscnt 0x0" ::: "memory");

    // ================= masked leaky-relu scores + row softmax (lane == row) =================
    {
        float f2i = sF2[wv][lane];
        float s[NNODE];
        float mx = -3.0e38f;
#pragma unroll
        for (int j = 0; j < NNODE; ++j) {
            float aij = sAdj[lane * NNODE + j];
            float lg  = (f2i + sF3[wv][j]) * aij;
            float sc  = (lg >= 0.f ? lg : 0.01f * lg) + (-1.0e9f) * (1.f - aij);
            s[j] = sc;
            mx = sc > mx ? sc : mx;
        }
        float sum = 0.f;
#pragma unroll
        for (int j = 0; j < NNODE; ++j) { float e = __expf(s[j] - mx); s[j] = e; sum += e; }
        float inv = 1.f / sum;
        half_t* aw = &sAw[wv][lane * NNODE];
#pragma unroll
        for (int j = 0; j < NNODE; ++j) aw[j] = (half_t)(s[j] * inv);
    }
    asm volatile("s_wait_dscnt 0x0" ::: "memory");

    // ================= att = att_w[32x32] @ x[32x64] + x  (residual), written back over x =================
    v8f attAcc[2][4];
#pragma unroll
    for (int mi = 0; mi < 2; ++mi)
#pragma unroll
        for (int ni = 0; ni < 4; ++ni) {
            v16h a  = ldsA(&sAw[wv][0], NNODE, mi * 16, 0, lane);
            v16h bb = ldsB_kn(xbuf, HDIM, ni * 16, 0, lane);
            attAcc[mi][ni] = wmma_f16(a, bb, z8);
        }
#pragma unroll
    for (int mi = 0; mi < 2; ++mi) {
        int r0 = mi * 16 + ((lane & 16) ? 8 : 0);
#pragma unroll
        for (int ni = 0; ni < 4; ++ni) {
            int col = ni * 16 + (lane & 15);
            float av[8];
#pragma unroll
            for (int r = 0; r < 8; ++r)
                av[r] = attAcc[mi][ni][r] + (float)xbuf[(r0 + r) * HDIM + col];
#pragma unroll
            for (int r = 0; r < 8; ++r)
                xbuf[(r0 + r) * HDIM + col] = (half_t)av[r];
        }
    }
    asm volatile("s_wait_dscnt 0x0" ::: "memory");

    // ================= GRUCell: gates via fused a@Wih^T and h@Whh^T =================
    const half_t* hbuf = &sH[wv][0];
#pragma unroll
    for (int mi = 0; mi < 2; ++mi) {
        v8f accR[4], accZ[4], accXN[4], accHN[4];
#pragma unroll
        for (int ni = 0; ni < 4; ++ni) { accR[ni] = z8; accZ[ni] = z8; accXN[ni] = z8; accHN[ni] = z8; }
#pragma unroll
        for (int kk = 0; kk < 2; ++kk) {
            v16h aA = ldsA(xbuf, HDIM, mi * 16, kk * 32, lane);   // a
            v16h aH = ldsA(hbuf, HDIM, mi * 16, kk * 32, lane);   // h_in
#pragma unroll
            for (int ni = 0; ni < 4; ++ni) {
                int n0 = ni * 16;
                accR[ni]  = wmma_f16(aA, gB_W(Wih, HDIM, 0 * HDIM + n0, kk * 32, lane), accR[ni]);
                accR[ni]  = wmma_f16(aH, gB_W(Whh, HDIM, 0 * HDIM + n0, kk * 32, lane), accR[ni]);
                accZ[ni]  = wmma_f16(aA, gB_W(Wih, HDIM, 1 * HDIM + n0, kk * 32, lane), accZ[ni]);
                accZ[ni]  = wmma_f16(aH, gB_W(Whh, HDIM, 1 * HDIM + n0, kk * 32, lane), accZ[ni]);
                accXN[ni] = wmma_f16(aA, gB_W(Wih, HDIM, 2 * HDIM + n0, kk * 32, lane), accXN[ni]);
                accHN[ni] = wmma_f16(aH, gB_W(Whh, HDIM, 2 * HDIM + n0, kk * 32, lane), accHN[ni]);
            }
        }
        int r0 = mi * 16 + ((lane & 16) ? 8 : 0);
#pragma unroll
        for (int ni = 0; ni < 4; ++ni) {
            int col = ni * 16 + (lane & 15);
            float biR  = bih[col] + bhh[col];
            float biZ  = bih[HDIM + col] + bhh[HDIM + col];
            float biXN = bih[2 * HDIM + col];
            float biHN = bhh[2 * HDIM + col];
#pragma unroll
            for (int r = 0; r < 8; ++r) {
                int row  = r0 + r;
                float rg = 1.f / (1.f + __expf(-(accR[ni][r] + biR)));
                float zg = 1.f / (1.f + __expf(-(accZ[ni][r] + biZ)));
                float ng = tanhf(accXN[ni][r] + biXN + rg * (accHN[ni][r] + biHN));
                float hin = hidden[(b * NNODE + row) * HDIM + col];
                float hv  = (1.f - zg) * ng + zg * hin;
                outh[(b * NNODE + row) * HDIM + col] = hv;
                sH[wv][row * HDIM + col] = (half_t)hv;     // rows of this m-tile only
            }
        }
    }
    asm volatile("s_wait_dscnt 0x0" ::: "memory");

    // ================= q = h @ Wq^T + bq =================
#pragma unroll
    for (int mi = 0; mi < 2; ++mi) {
        v8f acc = z8;
#pragma unroll
        for (int kk = 0; kk < 2; ++kk) {
            v16h aH = ldsA(hbuf, HDIM, mi * 16, kk * 32, lane);
            acc = wmma_f16(aH, gB_W(Wq, HDIM, 0, kk * 32, lane), acc);
        }
        int col = lane & 15;
        int r0  = mi * 16 + ((lane & 16) ? 8 : 0);
        float bqv = bq[col];
#pragma unroll
        for (int r = 0; r < 8; ++r)
            outq[(b * NNODE + r0 + r) * (long long)ADIM + col] = acc[r] + bqv;
    }
}

extern "C" void kernel_launch(void* const* d_in, const int* in_sizes, int n_in,
                              void* d_out, int out_size, void* d_ws, size_t ws_size,
                              hipStream_t stream) {
    const float* inp    = (const float*)d_in[0];
    const float* hidden = (const float*)d_in[1];
    const float* adj    = (const float*)d_in[2];
    const float* w1     = (const float*)d_in[3];
    const float* b1     = (const float*)d_in[4];
    const float* w2     = (const float*)d_in[5];
    const float* b2     = (const float*)d_in[6];
    const float* w3     = (const float*)d_in[7];
    const float* b3     = (const float*)d_in[8];
    const float* Wih    = (const float*)d_in[9];
    const float* Whh    = (const float*)d_in[10];
    const float* bih    = (const float*)d_in[11];
    const float* bhh    = (const float*)d_in[12];
    const float* Wq     = (const float*)d_in[13];
    const float* bq     = (const float*)d_in[14];

    float* outq = (float*)d_out;
    float* outh = outq + (long long)BATCH * NNODE * ADIM;

    dim3 grid(BATCH / WAVES);
    dim3 block(WAVES * 32);
    gat_gru_fused<<<grid, block, 0, stream>>>(inp, hidden, adj, w1, b1, w2, b2, w3, b3,
                                              Wih, Whh, bih, bhh, Wq, bq, outq, outh);
}
